// LoRAExpertAttention_4131758539047
// MI455X (gfx1250) — compile-verified, matrix operand order checked by reading, not run on hardware
//
#include <hip/hip_runtime.h>
#include <hip/hip_bf16.h>
#include <math.h>

// ---------------- problem constants ----------------
constexpr int B_    = 4;
constexpr int S_    = 1024;
constexpr int D_    = 2048;
constexpr int NH_   = 16;
constexpr int NKV_  = 8;
constexpr int HD_   = 128;
constexpr int NEXP_ = 4;
constexpr int RANK_ = 16;
constexpr int TOK_  = B_ * S_;          // 4096 tokens
constexpr int QDIM_ = NH_ * HD_;        // 2048
constexpr int KVDIM_= NKV_ * HD_;       // 1024
constexpr float LORA_SCALE_ = 2.0f;     // ALPHA / RANK = 32/16
constexpr float EPS_     = 1e-6f;
constexpr float NEG_INF_ = -1e9f;
constexpr float SCALING_ = 0.08838834764831845f; // HD^-0.5

typedef __bf16 bf16_t;
typedef __attribute__((ext_vector_type(16))) __bf16 v16bf;
typedef __attribute__((ext_vector_type(8)))  __bf16 v8bf;
typedef __attribute__((ext_vector_type(8)))  float  v8f;

union FragU {
    v16bf v;
    v8bf  h[2];
    bf16_t e[16];
};

// Load a 16-bf16 WMMA A/B fragment whose 16 values are two contiguous runs of 8:
//   elements 0..7  at p[0..7]      (K = k0 + half*8 .. +7)
//   elements 8..15 at p[16..23]    (K = k0 + 16 + half*8 .. +7)
// caller passes p = rowBase + k0 + half*8  (16-byte aligned)
__device__ inline v16bf frag_load(const bf16_t* p) {
    FragU f;
    f.h[0] = *(const v8bf*)(p);
    f.h[1] = *(const v8bf*)(p + 16);
    return f.v;
}

// ---- CDNA5 async global->LDS copy (ASYNCcnt path), with safe fallbacks ----
typedef int v4i_ __attribute__((vector_size(16)));
typedef __attribute__((address_space(1))) v4i_* as1_v4i_ptr;
typedef __attribute__((address_space(3))) v4i_* as3_v4i_ptr;

__device__ inline void async_copy16(const bf16_t* g, bf16_t* l) {
#if defined(__AMDGCN__) && __has_builtin(__builtin_amdgcn_global_load_async_to_lds_b128)
    __builtin_amdgcn_global_load_async_to_lds_b128(
        (as1_v4i_ptr)g, (as3_v4i_ptr)l, 0, 0);
#else
    *(v8bf*)l = *(const v8bf*)g;   // fallback: load + ds_store
#endif
}
__device__ inline void async_wait0() {
#if defined(__AMDGCN__) && __has_builtin(__builtin_amdgcn_global_load_async_to_lds_b128)
#if __has_builtin(__builtin_amdgcn_s_wait_asynccnt)
    __builtin_amdgcn_s_wait_asynccnt(0);
#else
    asm volatile("s_wait_asynccnt 0x0" ::: "memory");
#endif
#endif
}

__device__ inline float wave_reduce_add(float v) {
    #pragma unroll
    for (int off = 16; off > 0; off >>= 1) v += __shfl_xor(v, off, 32);
    return v;
}
__device__ inline float wave_reduce_max(float v) {
    #pragma unroll
    for (int off = 16; off > 0; off >>= 1) v = fmaxf(v, __shfl_xor(v, off, 32));
    return v;
}

// ---------------- 1) f32 -> bf16 conversion ----------------
__global__ void k_f32_to_bf16(const float* __restrict__ in, bf16_t* __restrict__ out, int n) {
    int i = blockIdx.x * blockDim.x + threadIdx.x;
    if (i < n) out[i] = (bf16_t)in[i];
}

// ---------------- 2) router gating: softmax + top-2 ----------------
__global__ void k_gate_topk(const float* __restrict__ x,   // [TOK, D]
                            const float* __restrict__ gw,  // [NEXP, D]
                            float* __restrict__ ew)        // [TOK, NEXP]
{
    int wid  = (blockIdx.x * blockDim.x + threadIdx.x) >> 5;
    int lane = threadIdx.x & 31;
    if (wid >= TOK_) return;
    const float* xr = x + (size_t)wid * D_;
    float acc[NEXP_] = {0.f, 0.f, 0.f, 0.f};
    for (int d = lane; d < D_; d += 32) {
        float xv = xr[d];
        #pragma unroll
        for (int e = 0; e < NEXP_; e++) acc[e] += xv * gw[e * D_ + d];
    }
    #pragma unroll
    for (int e = 0; e < NEXP_; e++) acc[e] = wave_reduce_add(acc[e]);

    float m = fmaxf(fmaxf(acc[0], acc[1]), fmaxf(acc[2], acc[3]));
    float p[NEXP_]; float s = 0.f;
    #pragma unroll
    for (int e = 0; e < NEXP_; e++) { p[e] = __expf(acc[e] - m); s += p[e]; }
    float inv = 1.0f / s;
    #pragma unroll
    for (int e = 0; e < NEXP_; e++) p[e] *= inv;

    int i1 = 0;
    #pragma unroll
    for (int e = 1; e < NEXP_; e++) if (p[e] > p[i1]) i1 = e;
    int i2 = -1;
    #pragma unroll
    for (int e = 0; e < NEXP_; e++)
        if (e != i1 && (i2 < 0 || p[e] > p[i2])) i2 = e;

    if (lane == 0) {
        #pragma unroll
        for (int e = 0; e < NEXP_; e++)
            ew[(size_t)wid * NEXP_ + e] = (e == i1 || e == i2) ? p[e] : 0.f;
    }
}

// ---------------- 3) LDS-tiled bf16 WMMA GEMM: C[M,N] = A[M,K] * W[N,K]^T ----------
// Block: 256 threads = 8 waves (4 M x 2 N). Macro tile 128x64.
// Each wave: 2x2 grid of 16x16 WMMA tiles (32x32 per wave).
// Per K-step of 32: A panel 128x32 (8KB) + B panel 64x32 (4KB) staged into LDS
// via GLOBAL_LOAD_ASYNC_TO_LDS_B128 (ASYNCcnt), then ds_load_b128 fragments.
constexpr int BM_ = 128, BN_ = 64, BK_ = 32;

__global__ __launch_bounds__(256)
void k_gemm_bf16_tiled(const bf16_t* __restrict__ A,   // [M, K]
                       const bf16_t* __restrict__ Bw,  // [N, K]
                       float* __restrict__ C,          // [M, N]
                       int M, int N, int K)
{
    __shared__ bf16_t Alds[BM_ * BK_];   // 8 KB, row r chunk j at byte r*64 + j*16
    __shared__ bf16_t Blds[BN_ * BK_];   // 4 KB

    const int tid  = threadIdx.x;
    const int lane = tid & 31;
    const int wave = tid >> 5;
    const int rn   = lane & 15, half = lane >> 4;
    const int wm   = wave >> 1;          // 0..3  (M sub-block of 32 rows)
    const int wn   = wave & 1;           // 0..1  (N sub-block of 32 cols)

    const int nbN = N / BN_;
    const int tm0 = (blockIdx.x / nbN) * BM_;
    const int tn0 = (blockIdx.x % nbN) * BN_;

    const bf16_t* Abase = A  + (size_t)tm0 * K;
    const bf16_t* Bbase = Bw + (size_t)tn0 * K;

    v8f acc[2][2];
    #pragma unroll
    for (int a = 0; a < 2; a++)
        #pragma unroll
        for (int b = 0; b < 2; b++)
            acc[a][b] = (v8f){0.f,0.f,0.f,0.f,0.f,0.f,0.f,0.f};

    for (int k0 = 0; k0 < K; k0 += BK_) {
        // ---- stage panels: A = 512 16B-chunks, B = 256 16B-chunks ----
        {
            int c0 = tid, c1 = tid + 256;
            async_copy16(Abase + (size_t)(c0 >> 2) * K + k0 + (c0 & 3) * 8, &Alds[c0 * 8]);
            async_copy16(Abase + (size_t)(c1 >> 2) * K + k0 + (c1 & 3) * 8, &Alds[c1 * 8]);
            async_copy16(Bbase + (size_t)(tid >> 2) * K + k0 + (tid & 3) * 8, &Blds[tid * 8]);
            async_wait0();
        }
        __syncthreads();

        // ---- fragments from LDS, 4 WMMAs per wave ----
        v16bf af[2], bf2[2];
        #pragma unroll
        for (int a = 0; a < 2; a++) {
            int row = wm * 32 + a * 16 + rn;
            af[a] = frag_load(&Alds[row * BK_ + half * 8]);
        }
        #pragma unroll
        for (int b = 0; b < 2; b++) {
            int col = wn * 32 + b * 16 + rn;
            bf2[b] = frag_load(&Blds[col * BK_ + half * 8]);
        }
        #pragma unroll
        for (int a = 0; a < 2; a++)
            #pragma unroll
            for (int b = 0; b < 2; b++)
                acc[a][b] = __builtin_amdgcn_wmma_f32_16x16x32_bf16(
                                false, af[a], false, bf2[b], (short)0, acc[a][b], false, false);
        __syncthreads();
    }

    // ---- write back: VGPR i -> row +half*8+i, col +rn ----
    #pragma unroll
    for (int a = 0; a < 2; a++) {
        #pragma unroll
        for (int b = 0; b < 2; b++) {
            float* crow = C + (size_t)(tm0 + wm * 32 + a * 16 + half * 8) * N
                            + tn0 + wn * 32 + b * 16 + rn;
            #pragma unroll
            for (int i = 0; i < 8; i++) crow[(size_t)i * N] = acc[a][b][i];
        }
    }
}

// ---------------- 4) fused RMSNorm + RoPE, write bf16 in [B, nh, S, HD] ----------------
__global__ void k_rmsnorm_rope(const float* __restrict__ inTok,  // [TOK, nh*HD]
                               const float* __restrict__ cosT,   // [TOK, HD]
                               const float* __restrict__ sinT,   // [TOK, HD]
                               const float* __restrict__ normw,  // [HD]
                               bf16_t* __restrict__ outBH,       // [B, nh, S, HD]
                               int nh)
{
    int wid  = (blockIdx.x * blockDim.x + threadIdx.x) >> 5;
    int lane = threadIdx.x & 31;
    int total = TOK_ * nh;
    if (wid >= total) return;
    int tok = wid / nh, h = wid % nh;
    int b = tok / S_, s = tok % S_;

    const float* xr = inTok + (size_t)tok * (nh * HD_) + h * HD_;
    float x[4];
    #pragma unroll
    for (int j = 0; j < 4; j++) x[j] = xr[lane + 32 * j];

    float ss = x[0]*x[0] + x[1]*x[1] + x[2]*x[2] + x[3]*x[3];
    ss = wave_reduce_add(ss);
    float r = rsqrtf(ss * (1.0f / HD_) + EPS_);

    float xn[4];
    #pragma unroll
    for (int j = 0; j < 4; j++) xn[j] = x[j] * r * normw[lane + 32 * j];

    const float* cr = cosT + (size_t)tok * HD_;
    const float* sr = sinT + (size_t)tok * HD_;
    // hd = lane + 32j ; rotate_half pairs (j=0,j=2) and (j=1,j=3) live in the same lane
    float o[4];
    o[0] = xn[0] * cr[lane]      + (-xn[2]) * sr[lane];
    o[1] = xn[1] * cr[lane + 32] + (-xn[3]) * sr[lane + 32];
    o[2] = xn[2] * cr[lane + 64] + ( xn[0]) * sr[lane + 64];
    o[3] = xn[3] * cr[lane + 96] + ( xn[1]) * sr[lane + 96];

    bf16_t* out = outBH + ((size_t)(b * nh + h) * S_ + s) * HD_;
    #pragma unroll
    for (int j = 0; j < 4; j++) out[lane + 32 * j] = (bf16_t)o[j];
}

// ---------------- 5) V: [TOK, NKV*HD] f32 -> [B, NKV, S, HD] bf16 ----------------
__global__ void k_v_transpose(const float* __restrict__ inTok, bf16_t* __restrict__ outBH)
{
    int wid  = (blockIdx.x * blockDim.x + threadIdx.x) >> 5;
    int lane = threadIdx.x & 31;
    int total = TOK_ * NKV_;
    if (wid >= total) return;
    int tok = wid / NKV_, kvh = wid % NKV_;
    int b = tok / S_, s = tok % S_;
    const float* xr = inTok + (size_t)tok * KVDIM_ + kvh * HD_;
    bf16_t* out = outBH + ((size_t)(b * NKV_ + kvh) * S_ + s) * HD_;
    #pragma unroll
    for (int j = 0; j < 4; j++) out[lane + 32 * j] = (bf16_t)xr[lane + 32 * j];
}

// ---------------- 6) causal attention, one wave per (b,h,q-block-of-16) ----------------
__global__ __launch_bounds__(32)
void k_attention(const bf16_t* __restrict__ Q,   // [B, NH, S, HD]
                 const bf16_t* __restrict__ Kt,  // [B, NKV, S, HD]
                 const bf16_t* __restrict__ V,   // [B, NKV, S, HD]
                 bf16_t* __restrict__ O)         // [TOK, NH*HD] token-major
{
    __shared__ float sbuf[16][S_];   // 64 KB score/prob buffer

    int bh = blockIdx.x;             // b*NH + h
    int qb = blockIdx.y;             // block of 16 query rows
    int b = bh / NH_, h = bh % NH_;
    int kvh = h / (NH_ / NKV_);      // GQA repeat = 2
    int lane = threadIdx.x & 31;
    int rn = lane & 15, half = lane >> 4;

    const bf16_t* Qbase = Q  + ((size_t)(b * NH_  + h  ) * S_) * HD_;
    const bf16_t* Kbase = Kt + ((size_t)(b * NKV_ + kvh) * S_) * HD_;
    const bf16_t* Vbase = V  + ((size_t)(b * NKV_ + kvh) * S_) * HD_;

    // Q fragments: A-matrix rows = q rows, K-dim = head_dim in 4 chunks of 32
    v16bf aq[4];
    {
        const bf16_t* qrow = Qbase + (size_t)(qb * 16 + rn) * HD_ + half * 8;
        #pragma unroll
        for (int c = 0; c < 4; c++) aq[c] = frag_load(qrow + c * 32);
    }

    const int L = (qb + 1) * 16;             // causal key extent for this block
    const int Lpad = (L + 31) & ~31;

    // ---- pass 1: S = scale * Q K^T, causal-masked, into LDS ----
    for (int kb = 0; kb <= qb; kb++) {
        v8f st = {0.f, 0.f, 0.f, 0.f, 0.f, 0.f, 0.f, 0.f};
        const bf16_t* krow = Kbase + (size_t)(kb * 16 + rn) * HD_ + half * 8;
        #pragma unroll
        for (int c = 0; c < 4; c++) {
            v16bf bk = frag_load(krow + c * 32);
            st = __builtin_amdgcn_wmma_f32_16x16x32_bf16(
                     false, aq[c], false, bk, (short)0, st, false, false);
        }
        int col = kb * 16 + rn;
        #pragma unroll
        for (int i = 0; i < 8; i++) {
            int row = half * 8 + i;
            int qg  = qb * 16 + row;
            sbuf[row][col] = (col <= qg) ? st[i] * SCALING_ : NEG_INF_;
        }
    }
    __syncthreads();

    // ---- softmax on each of the 16 rows (whole wave per row) ----
    for (int row = 0; row < 16; row++) {
        float m = -3.4e38f;
        for (int c = lane; c < L; c += 32) m = fmaxf(m, sbuf[row][c]);
        m = wave_reduce_max(m);
        float sum = 0.f;
        for (int c = lane; c < L; c += 32) {
            float e = __expf(sbuf[row][c] - m);
            sbuf[row][c] = e;
            sum += e;
        }
        sum = wave_reduce_add(sum);
        float inv = 1.0f / sum;
        for (int c = lane; c < Lpad; c += 32)
            sbuf[row][c] = (c < L) ? sbuf[row][c] * inv : 0.f;
    }
    __syncthreads();

    // ---- pass 2: O = P V  (P 16xL from LDS as bf16, V blocks of 32 keys) ----
    v8f acc[8];
    #pragma unroll
    for (int j = 0; j < 8; j++) acc[j] = (v8f){0.f,0.f,0.f,0.f,0.f,0.f,0.f,0.f};

    const int nkb = Lpad >> 5;
    for (int kb = 0; kb < nkb; kb++) {
        FragU pf;
        const float* prow = &sbuf[rn][kb * 32 + half * 8];
        #pragma unroll
        for (int j = 0; j < 8; j++) {
            pf.e[j]     = (bf16_t)prow[j];       // K = kb*32 + half*8 + j
            pf.e[8 + j] = (bf16_t)prow[16 + j];  // K = kb*32 + 16 + half*8 + j
        }
        #pragma unroll
        for (int jn = 0; jn < 8; jn++) {
            // B fragment: column hd = jn*16 + rn, rows = 32 keys (strided by HD)
            const bf16_t* vcol = Vbase + (size_t)(kb * 32 + half * 8) * HD_ + jn * 16 + rn;
            FragU vf;
            #pragma unroll
            for (int j = 0; j < 8; j++) {
                vf.e[j]     = vcol[(size_t)j * HD_];
                vf.e[8 + j] = vcol[(size_t)(16 + j) * HD_];
            }
            acc[jn] = __builtin_amdgcn_wmma_f32_16x16x32_bf16(
                          false, pf.v, false, vf.v, (short)0, acc[jn], false, false);
        }
    }

    // ---- write attn_out token-major [TOK, NH*HD] as bf16 ----
    #pragma unroll
    for (int jn = 0; jn < 8; jn++) {
        #pragma unroll
        for (int i = 0; i < 8; i++) {
            size_t token = (size_t)b * S_ + qb * 16 + half * 8 + i;
            O[token * QDIM_ + h * HD_ + jn * 16 + rn] = (bf16_t)acc[jn][i];
        }
    }
}

// ---------------- 7) LoRA h = x @ loraA^T  (per token, 64 (e,r) pairs) ----------------
__global__ void k_lora_h(const bf16_t* __restrict__ x,      // [TOK, QDIM]
                         const float* __restrict__ loraA,   // [NEXP*RANK, QDIM]
                         float* __restrict__ hbuf)          // [TOK, NEXP*RANK]
{
    int t    = blockIdx.x;
    int wave = threadIdx.x >> 5;
    int lane = threadIdx.x & 31;
    int nwaves = blockDim.x >> 5;
    const bf16_t* xr = x + (size_t)t * QDIM_;
    for (int p = wave; p < NEXP_ * RANK_; p += nwaves) {
        const float* ar = loraA + (size_t)p * QDIM_;
        float acc = 0.f;
        for (int d = lane; d < QDIM_; d += 32) acc += (float)xr[d] * ar[d];
        acc = wave_reduce_add(acc);
        if (lane == 0) hbuf[(size_t)t * (NEXP_ * RANK_) + p] = acc;
    }
}

// ---------------- 8) out += scale * sum_e ew_e * (h_e @ loraB_e^T) ----------------
__global__ void k_lora_delta(const float* __restrict__ hbuf,   // [TOK, NEXP*RANK]
                             const float* __restrict__ loraB,  // [NEXP, D, RANK]
                             const float* __restrict__ ew,     // [TOK, NEXP]
                             float* __restrict__ out)          // [TOK, D]
{
    int gid = blockIdx.x * blockDim.x + threadIdx.x;
    if (gid >= TOK_ * D_) return;
    int t = gid / D_, d = gid % D_;
    float acc = 0.f;
    #pragma unroll
    for (int e = 0; e < NEXP_; e++) {
        float w = ew[(size_t)t * NEXP_ + e];
        const float* br = loraB + ((size_t)e * D_ + d) * RANK_;
        const float* hr = hbuf + (size_t)t * (NEXP_ * RANK_) + e * RANK_;
        float s = 0.f;
        #pragma unroll
        for (int r = 0; r < RANK_; r++) s += hr[r] * br[r];
        acc += w * s;
    }
    out[gid] += acc * LORA_SCALE_;
}

// ---------------- host launch ----------------
extern "C" void kernel_launch(void* const* d_in, const int* in_sizes, int n_in,
                              void* d_out, int out_size, void* d_ws, size_t ws_size,
                              hipStream_t stream)
{
    const float* hidden = (const float*)d_in[0];   // [B,S,D]
    const float* cosT   = (const float*)d_in[1];   // [B,S,HD]
    const float* sinT   = (const float*)d_in[2];   // [B,S,HD]
    const float* Wq     = (const float*)d_in[3];   // [QDIM, D]
    const float* Wk     = (const float*)d_in[4];   // [KVDIM, D]
    const float* Wv     = (const float*)d_in[5];   // [KVDIM, D]
    const float* Wo     = (const float*)d_in[6];   // [D, QDIM]
    const float* qnw    = (const float*)d_in[7];   // [HD]
    const float* knw    = (const float*)d_in[8];   // [HD]
    const float* gatew  = (const float*)d_in[9];   // [NEXP, D]
    const float* loraA  = (const float*)d_in[10];  // [NEXP, RANK, QDIM]
    const float* loraB  = (const float*)d_in[11];  // [NEXP, D, RANK]
    float* out = (float*)d_out;                    // [TOK, D]

    // workspace bump allocator
    char* ws = (char*)d_ws;
    size_t off = 0;
    auto alloc = [&](size_t bytes) -> void* {
        void* p = ws + off;
        off += (bytes + 255) & ~(size_t)255;
        return p;
    };
    bf16_t* hbf   = (bf16_t*)alloc((size_t)TOK_ * D_ * 2);
    bf16_t* wqbf  = (bf16_t*)alloc((size_t)QDIM_ * D_ * 2);
    bf16_t* wkbf  = (bf16_t*)alloc((size_t)KVDIM_ * D_ * 2);
    bf16_t* wvbf  = (bf16_t*)alloc((size_t)KVDIM_ * D_ * 2);
    bf16_t* wobf  = (bf16_t*)alloc((size_t)D_ * QDIM_ * 2);
    float*  qf    = (float*)alloc((size_t)TOK_ * QDIM_ * 4);   // reused for attn_out bf16
    float*  kf    = (float*)alloc((size_t)TOK_ * KVDIM_ * 4);
    float*  vf    = (float*)alloc((size_t)TOK_ * KVDIM_ * 4);
    bf16_t* Qb    = (bf16_t*)alloc((size_t)B_ * NH_  * S_ * HD_ * 2);
    bf16_t* Kb    = (bf16_t*)alloc((size_t)B_ * NKV_ * S_ * HD_ * 2);
    bf16_t* Vb    = (bf16_t*)alloc((size_t)B_ * NKV_ * S_ * HD_ * 2);
    float*  ew    = (float*)alloc((size_t)TOK_ * NEXP_ * 4);
    float*  hlora = (float*)alloc((size_t)TOK_ * NEXP_ * RANK_ * 4);
    bf16_t* attnb = (bf16_t*)qf;  // alias: qf (32MB) is dead after RoPE; attn_out bf16 needs 16MB

    // 1) conversions
    {
        int n, blk = 256;
        n = TOK_ * D_;       k_f32_to_bf16<<<(n + blk - 1) / blk, blk, 0, stream>>>(hidden, hbf, n);
        n = QDIM_ * D_;      k_f32_to_bf16<<<(n + blk - 1) / blk, blk, 0, stream>>>(Wq, wqbf, n);
        n = KVDIM_ * D_;     k_f32_to_bf16<<<(n + blk - 1) / blk, blk, 0, stream>>>(Wk, wkbf, n);
        n = KVDIM_ * D_;     k_f32_to_bf16<<<(n + blk - 1) / blk, blk, 0, stream>>>(Wv, wvbf, n);
        n = D_ * QDIM_;      k_f32_to_bf16<<<(n + blk - 1) / blk, blk, 0, stream>>>(Wo, wobf, n);
    }

    // 2) gating (one wave per token)
    k_gate_topk<<<(TOK_ * 32 + 255) / 256, 256, 0, stream>>>(hidden, gatew, ew);

    // 3) QKV projections (LDS-tiled WMMA, 128x64 macro tiles)
    {
        int blocks;
        blocks = (TOK_ / BM_) * (QDIM_ / BN_);
        k_gemm_bf16_tiled<<<blocks, 256, 0, stream>>>(hbf, wqbf, qf, TOK_, QDIM_, D_);
        blocks = (TOK_ / BM_) * (KVDIM_ / BN_);
        k_gemm_bf16_tiled<<<blocks, 256, 0, stream>>>(hbf, wkbf, kf, TOK_, KVDIM_, D_);
        k_gemm_bf16_tiled<<<blocks, 256, 0, stream>>>(hbf, wvbf, vf, TOK_, KVDIM_, D_);
    }

    // 4) RMSNorm + RoPE (Q and K), V layout convert
    k_rmsnorm_rope<<<(TOK_ * NH_ * 32 + 255) / 256, 256, 0, stream>>>(qf, cosT, sinT, qnw, Qb, NH_);
    k_rmsnorm_rope<<<(TOK_ * NKV_ * 32 + 255) / 256, 256, 0, stream>>>(kf, cosT, sinT, knw, Kb, NKV_);
    k_v_transpose<<<(TOK_ * NKV_ * 32 + 255) / 256, 256, 0, stream>>>(vf, Vb);

    // 5) attention: grid (B*NH, S/16), one wave per block
    {
        dim3 grid(B_ * NH_, S_ / 16);
        k_attention<<<grid, 32, 0, stream>>>(Qb, Kb, Vb, attnb);
    }

    // 6) base output projection -> d_out
    {
        int blocks = (TOK_ / BM_) * (D_ / BN_);
        k_gemm_bf16_tiled<<<blocks, 256, 0, stream>>>(attnb, wobf, out, TOK_, D_, QDIM_);
    }

    // 7) LoRA
    k_lora_h<<<TOK_, 256, 0, stream>>>(attnb, loraA, hlora);
    k_lora_delta<<<(TOK_ * D_ + 255) / 256, 256, 0, stream>>>(hlora, loraB, ew, out);

    (void)in_sizes; (void)n_in; (void)out_size; (void)ws_size;
}
